// MultiTaskFEGIN_10127532884202
// MI455X (gfx1250) — matrix operand order repaired and checked by reading, not compiled, not run on hardware
//
#include <hip/hip_runtime.h>
#include <hip/hip_bf16.h>
#include <math.h>

typedef _Float16 half_t;
typedef __attribute__((ext_vector_type(16))) _Float16 v16h;
typedef __attribute__((ext_vector_type(8)))  float    v8f;

#define N_NODES   51200
#define N_GRAPHS  64
#define NPG       800
#define DEG       16
#define EPG       (NPG*DEG)      /* 12800 edges per graph */
#define N_EDGES   (N_NODES*DEG)
#define CANDS     6400
#define F         128
#define D_EMB     512            /* 4 layers * 128 */
#define BN_EPS    1e-5f

// ---------------------------------------------------------------- utilities
__global__ void k_copy_f32(const float* __restrict__ in, float* __restrict__ out, int n) {
  int i = blockIdx.x * blockDim.x + threadIdx.x;
  if (i < n) out[i] = in[i];
}

__global__ void k_zero_f32(float* __restrict__ p, int n) {
  int i = blockIdx.x * blockDim.x + threadIdx.x;
  if (i < n) p[i] = 0.0f;
}

// Weight pre-pass: W[K,N] f32 row-major  ->  Wt[Npad,K] f16 (transposed,
// zero-padded columns). Done once per launch; lets the GEMM pull B fragments
// as two contiguous global_load_b128 per k-step with zero guards.
__global__ void k_wt(const float* __restrict__ W, half_t* __restrict__ Wt,
                     int K, int N, int Npad) {
  int idx = blockIdx.x * blockDim.x + threadIdx.x;
  if (idx >= Npad * K) return;
  int n = idx / K;
  int k = idx - n * K;
  Wt[idx] = (n < N) ? (half_t)W[(long)k * N + n] : (half_t)0.0f;
}

// ------------------------------------------------- GIN neighbor aggregation
// One workgroup per (graph, feature-half). 800x64 fp32 tile in LDS (200 KB),
// scatter-add of 12800 in-graph edges via ds_add_f32 LDS atomics.
// Output: z = (1+eps)*h + sum_{j->i} h_j
__global__ void k_gin_aggregate(const float* __restrict__ h,
                                const int* __restrict__ src,
                                const int* __restrict__ dst,
                                const float* __restrict__ eps_p,
                                float* __restrict__ z) {
  extern __shared__ float lds[];            // NPG*64 floats
  const int gph   = blockIdx.x >> 1;
  const int half  = blockIdx.x & 1;
  const int fbase = half * 64;
  const int nbase = gph * NPG;
  const int tid   = threadIdx.x;            // 512 threads = 16 waves

  for (int i = tid; i < NPG * 64; i += 512) lds[i] = 0.0f;
  __syncthreads();

  const int fe    = tid & 63;               // feature within half
  const int eg    = tid >> 6;               // edge subgroup 0..7
  const int ebase = gph * EPG;
  for (int i = eg; i < EPG; i += 8) {
    const int e = ebase + i;
    const int s = src[e];
    const int d = dst[e] - nbase;
    atomicAdd(&lds[d * 64 + fe], h[s * F + fbase + fe]);  // ds_add_f32
  }
  __syncthreads();

  const float eps1 = 1.0f + eps_p[0];
  for (int i = tid; i < NPG * 64; i += 512) {
    const int node = i >> 6, f = i & 63;
    const int gi = (nbase + node) * F + fbase + f;
    z[gi] = eps1 * h[gi] + lds[i];
  }
}

// --------------------------------------------------------- generic WMMA GEMM
// C[M,N] = act(A[M,K] @ W[K,N] + bias).  A row-major (f32 or f16); weights
// pre-transposed/padded f16 Wt[Npad,K].  Block = 256 threads = 8 waves;
// block tile 16 (M) x 128 (N); each wave owns a 16x16 N-tile.  The whole
// 16xK A strip is staged to LDS once (f16), then the K loop fully unrolls:
// per step 2x ds_load_b128 (A frag) + 2x global_load_b128 (B frag) + 1 wmma.
template <int K, bool AF16>
__global__ void k_wmma_gemm(const void* __restrict__ Ap,
                            const half_t* __restrict__ Wt,
                            const float* __restrict__ bias,
                            float* __restrict__ C,
                            int N, int relu) {
  __shared__ half_t As[16 * K];

  const int tid  = threadIdx.x;     // 0..255
  const int lane = tid & 31;
  const int wave = tid >> 5;        // 0..7 -> 16-col slice
  const int mt   = blockIdx.x;      // 16-row tile
  const int nb   = blockIdx.y * 128;
  const int lrow = lane & 15;
  const int hi   = lane >> 4;       // 0 / 1 half of the wave

  // ---- stage full 16xK A strip into LDS as f16 (vectorized, no guards:
  //      all M are multiples of 16 and grid.x == M/16)
  constexpr int HK = K / 2;         // half-pairs per row
  for (int idx = tid; idx < 16 * HK; idx += 256) {
    const int r  = idx / HK;        // HK is a power of two -> shifts
    const int kp = idx & (HK - 1);
    if (AF16) {
      const unsigned v = *(const unsigned*)((const half_t*)Ap +
                          (long)(mt * 16 + r) * K + 2 * kp);
      *(unsigned*)&As[r * K + 2 * kp] = v;
    } else {
      const float* ap = (const float*)Ap + (long)(mt * 16 + r) * K + 2 * kp;
      As[r * K + 2 * kp]     = (half_t)ap[0];
      As[r * K + 2 * kp + 1] = (half_t)ap[1];
    }
  }
  __syncthreads();

  const int col = nb + wave * 16 + lrow;   // column in padded weight space
  v8f acc = {};
#pragma unroll
  for (int kb = 0; kb < K; kb += 32) {
    // A fragment: lane<16 -> row=lrow, K {kb+0..7, kb+16..23};
    //             lane>=16 ->           K {kb+8..15, kb+24..31}
    union { int4 q[2]; v16h v; } au;
    au.q[0] = *(const int4*)&As[lrow * K + kb + hi * 8];
    au.q[1] = *(const int4*)&As[lrow * K + kb + 16 + hi * 8];
    // B fragment: lane<16 -> col, K {kb+0..7, kb+16..23}; lane>=16 -> K {kb+8..15, kb+24..31}
    union { int4 q[2]; v16h v; } bu;
    const half_t* wp = Wt + (long)col * K + kb + hi * 8;
    bu.q[0] = *(const int4*)wp;
    bu.q[1] = *(const int4*)(wp + 16);
    acc = __builtin_amdgcn_wmma_f32_16x16x32_f16(
        false, au.v, false, bu.v, (short)0, acc, false, false);
  }

  // epilogue: D VGPR r -> M=r (lanes 0-15) / M=r+8 (lanes 16-31), N=lane%16
  if (col < N) {
    const float bval = bias ? bias[col] : 0.0f;
#pragma unroll
    for (int r = 0; r < 8; ++r) {
      const int m = mt * 16 + r + hi * 8;
      float v = acc[r] + bval;
      if (relu) v = fmaxf(v, 0.0f);
      C[(long)m * N + col] = v;
    }
  }
}

// ------------------------------------------------------------ BatchNorm1d
__global__ void k_bn_stats(const float* __restrict__ z, float* __restrict__ stats) {
  const int c  = threadIdx.x;               // 128 threads = feature columns
  const int r0 = blockIdx.x * 256;
  float s = 0.0f, s2 = 0.0f;
  const int rend = min(r0 + 256, N_NODES);
  for (int r = r0; r < rend; ++r) {
    const float v = z[(long)r * F + c];
    s += v; s2 += v * v;
  }
  atomicAdd(&stats[c], s);
  atomicAdd(&stats[F + c], s2);
}

__global__ void k_bn_apply(const float* __restrict__ z, const float* __restrict__ stats,
                           const float* __restrict__ gamma, const float* __restrict__ beta,
                           float* __restrict__ hout, half_t* __restrict__ emb, int layer) {
  const int idx = blockIdx.x * blockDim.x + threadIdx.x;
  if (idx >= N_NODES * F) return;
  const int c   = idx & (F - 1);
  const int row = idx >> 7;
  const float inv_n = 1.0f / (float)N_NODES;
  const float mu  = stats[c] * inv_n;
  const float var = stats[F + c] * inv_n - mu * mu;
  const float v = gamma[c] * (z[idx] - mu) * rsqrtf(var + BN_EPS) + beta[c];
  hout[idx] = v;
  emb[(long)row * D_EMB + layer * F + c] = (half_t)v;
}

// -------------------------------------------------------------- graph pool
__global__ void k_pool(const half_t* __restrict__ emb, half_t* __restrict__ g16) {
  const int gph = blockIdx.x;               // 64 graphs
  for (int c = threadIdx.x; c < D_EMB; c += 256) {
    float s = 0.0f;
    for (int r = 0; r < NPG; ++r)
      s += (float)emb[(long)(gph * NPG + r) * D_EMB + c];
    g16[gph * D_EMB + c] = (half_t)(s * (1.0f / NPG));
  }
}

__global__ void k_log_softmax(const float* __restrict__ logits, float* __restrict__ out) {
  const int row = blockIdx.x * blockDim.x + threadIdx.x;
  if (row >= N_GRAPHS) return;
  const float* l = logits + row * 16;
  float mx = l[0];
  for (int i = 1; i < 16; ++i) mx = fmaxf(mx, l[i]);
  float s = 0.0f;
  for (int i = 0; i < 16; ++i) s += __expf(l[i] - mx);
  const float lse = mx + __logf(s);
  for (int i = 0; i < 16; ++i) out[row * 16 + i] = l[i] - lse;
}

// -------------------------------------------------------------- link pred
__global__ void k_gather(const half_t* __restrict__ emb, const int* __restrict__ cs,
                         const int* __restrict__ cd, half_t* __restrict__ feats) {
  const int idx = blockIdx.x * blockDim.x + threadIdx.x;
  if (idx >= CANDS * 1024) return;
  const int e = idx >> 10, c = idx & 1023;
  const int node = (c < 512) ? cs[e] : cd[e];
  feats[idx] = emb[(long)node * D_EMB + (c & 511)];
}

__global__ void k_edge_score(const float* __restrict__ e1, const float* __restrict__ W2,
                             const float* __restrict__ b2, float* __restrict__ out) {
  const int lane = threadIdx.x & 31;
  const int wave = threadIdx.x >> 5;
  const int e = blockIdx.x * 8 + wave;
  if (e >= CANDS) return;
  float s = 0.0f;
  for (int j = lane; j < 128; j += 32) s += e1[(long)e * 128 + j] * W2[j];
  for (int ofs = 16; ofs > 0; ofs >>= 1) s += __shfl_xor(s, ofs, 32);
  if (lane == 0) out[e] = 1.0f / (1.0f + __expf(-(s + b2[0])));
}

// ----------------------------------------------------------------- launch
extern "C" void kernel_launch(void* const* d_in, const int* in_sizes, int n_in,
                              void* d_out, int out_size, void* d_ws, size_t ws_size,
                              hipStream_t stream) {
  const float* x   = (const float*)d_in[0];
  const int*   src = (const int*)d_in[1];
  const int*   dst = src + N_EDGES;
  const int*   cs  = (const int*)d_in[3];
  const int*   cd  = cs + CANDS;
  // params flattened in dict insertion order after the 4 arrays:
  // layer l: [4+7l]=W1 b1 W2 b2 bng bnb eps ; nc: 32..39 ; ep: 40..45
  auto P = [&](int i) { return (const float*)d_in[i]; };

  char* ws = (char*)d_ws;
  size_t off = 0;
  auto carve = [&](size_t bytes) -> void* {
    void* p = ws + off;
    off = (off + bytes + 255) & ~(size_t)255;
    return p;
  };
  float*  h     = (float*)carve((size_t)N_NODES * F * 4);
  float*  z     = (float*)carve((size_t)N_NODES * F * 4);
  float*  t     = (float*)carve((size_t)N_NODES * F * 4);
  half_t* emb   = (half_t*)carve((size_t)N_NODES * D_EMB * 2);
  float*  stats = (float*)carve(256 * 4);
  half_t* g16   = (half_t*)carve((size_t)N_GRAPHS * D_EMB * 2);
  float*  nc0   = (float*)carve((size_t)N_GRAPHS * 256 * 4);
  float*  nc1   = (float*)carve((size_t)N_GRAPHS * 128 * 4);
  float*  nc2   = (float*)carve((size_t)N_GRAPHS * 128 * 4);
  float*  logit = (float*)carve((size_t)N_GRAPHS * 16 * 4);
  half_t* feats = (half_t*)carve((size_t)CANDS * 1024 * 2);
  float*  e0    = (float*)carve((size_t)CANDS * 256 * 4);
  float*  e1b   = (float*)carve((size_t)CANDS * 128 * 4);
  // transposed/padded f16 weights: Wt[Npad][K]
  half_t* wt_c[4][2];
  for (int l = 0; l < 4; ++l) {
    wt_c[l][0] = (half_t*)carve((size_t)128 * 128 * 2);
    wt_c[l][1] = (half_t*)carve((size_t)128 * 128 * 2);
  }
  half_t* wt_nc0 = (half_t*)carve((size_t)256 * 512 * 2);
  half_t* wt_nc1 = (half_t*)carve((size_t)128 * 256 * 2);
  half_t* wt_nc2 = (half_t*)carve((size_t)128 * 128 * 2);
  half_t* wt_nc3 = (half_t*)carve((size_t)128 * 128 * 2);   // N=16 padded
  half_t* wt_ep0 = (half_t*)carve((size_t)256 * 1024 * 2);
  half_t* wt_ep1 = (half_t*)carve((size_t)128 * 256 * 2);
  (void)ws_size; (void)n_in; (void)in_sizes; (void)out_size;

  auto wt = [&](const float* W, half_t* Wt, int K, int N, int Npad) {
    k_wt<<<(Npad * K + 255) / 256, 256, 0, stream>>>(W, Wt, K, N, Npad);
  };
  for (int l = 0; l < 4; ++l) {
    const int pb = 4 + 7 * l;
    wt(P(pb + 0), wt_c[l][0], 128, 128, 128);
    wt(P(pb + 2), wt_c[l][1], 128, 128, 128);
  }
  wt(P(32), wt_nc0, 512, 256, 256);
  wt(P(34), wt_nc1, 256, 128, 128);
  wt(P(36), wt_nc2, 128, 128, 128);
  wt(P(38), wt_nc3, 128, 16, 128);
  wt(P(40), wt_ep0, 1024, 256, 256);
  wt(P(42), wt_ep1, 256, 128, 128);

  k_copy_f32<<<(N_NODES * F + 255) / 256, 256, 0, stream>>>(x, h, N_NODES * F);

  for (int l = 0; l < 4; ++l) {
    const int pb = 4 + 7 * l;
    k_gin_aggregate<<<N_GRAPHS * 2, 512, NPG * 64 * sizeof(float), stream>>>(
        h, src, dst, P(pb + 6), z);
    k_wmma_gemm<128, false><<<dim3(N_NODES / 16, 1), 256, 0, stream>>>(
        z, wt_c[l][0], P(pb + 1), t, F, 1);
    k_wmma_gemm<128, false><<<dim3(N_NODES / 16, 1), 256, 0, stream>>>(
        t, wt_c[l][1], P(pb + 3), z, F, 1);
    k_zero_f32<<<1, 256, 0, stream>>>(stats, 256);
    k_bn_stats<<<N_NODES / 256, 128, 0, stream>>>(z, stats);
    k_bn_apply<<<(N_NODES * F + 255) / 256, 256, 0, stream>>>(
        z, stats, P(pb + 4), P(pb + 5), h, emb, l);
  }

  // graph classifier head
  k_pool<<<N_GRAPHS, 256, 0, stream>>>(emb, g16);
  k_wmma_gemm<512, true><<<dim3(N_GRAPHS / 16, 2), 256, 0, stream>>>(
      g16, wt_nc0, P(33), nc0, 256, 1);
  k_wmma_gemm<256, false><<<dim3(N_GRAPHS / 16, 1), 256, 0, stream>>>(
      nc0, wt_nc1, P(35), nc1, 128, 1);
  k_wmma_gemm<128, false><<<dim3(N_GRAPHS / 16, 1), 256, 0, stream>>>(
      nc1, wt_nc2, P(37), nc2, 128, 1);
  k_wmma_gemm<128, false><<<dim3(N_GRAPHS / 16, 1), 256, 0, stream>>>(
      nc2, wt_nc3, P(39), logit, 16, 0);
  k_log_softmax<<<1, 64, 0, stream>>>(logit, (float*)d_out);

  // link predictor head
  k_gather<<<(CANDS * 1024 + 255) / 256, 256, 0, stream>>>(emb, cs, cd, feats);
  k_wmma_gemm<1024, true><<<dim3(CANDS / 16, 2), 256, 0, stream>>>(
      feats, wt_ep0, P(41), e0, 256, 1);
  k_wmma_gemm<256, false><<<dim3(CANDS / 16, 1), 256, 0, stream>>>(
      e0, wt_ep1, P(43), e1b, 128, 1);
  k_edge_score<<<CANDS / 8, 256, 0, stream>>>(e1b, P(44), P(45), (float*)d_out + N_GRAPHS * 16);
}
